// MyRNN_18442589569703
// MI455X (gfx1250) — matrix operand order repaired and checked by reading, not compiled, not run on hardware
//
#include <hip/hip_runtime.h>

// ---------------------------------------------------------------------------
// Stacked SimpleRNN forward on gfx1250 (MI455X), bf16 WMMA with f32 accum.
//   tokens[512,80] -> emb[10000,100] -> 2x RNN(2048) over T=80 -> head -> sigmoid
// Pipelined double-buffered LDS GEMM, async-to-LDS (GLOBAL_LOAD_ASYNC_TO_LDS_B128).
// ---------------------------------------------------------------------------

typedef __bf16 bf16_t;
typedef bf16_t v16bf __attribute__((ext_vector_type(16)));
typedef bf16_t v8bf  __attribute__((ext_vector_type(8)));
typedef float  v8f   __attribute__((ext_vector_type(8)));
typedef int    v4i   __attribute__((ext_vector_type(4)));

#define BATCH 512
#define SEQ   80
#define EMB   100
#define EMBP  128      // padded K for layer-0 input term
#define UNITS 2048
#define KC    32       // K per WMMA (bf16)
#define KSTEP 64       // K staged per barrier (2 WMMA chunks)
#define BM    128      // block tile M
#define BN    64       // block tile N
#define LDS_K 72       // LDS row stride (elems) for 64-K stage, conflict pad

#if defined(__has_builtin)
#if __has_builtin(__builtin_amdgcn_global_load_async_to_lds_b128)
#define HAVE_ASYNC_LDS 1
#endif
#endif

__device__ __forceinline__ bf16_t f2bf(float f) {
  unsigned int u = __builtin_bit_cast(unsigned int, f);
  unsigned int r = u + 0x7FFFu + ((u >> 16) & 1u);   // round-to-nearest-even
  unsigned short h = (unsigned short)(r >> 16);
  return __builtin_bit_cast(bf16_t, h);
}
__device__ __forceinline__ float bf2f(bf16_t b) {
  unsigned short h = __builtin_bit_cast(unsigned short, b);
  unsigned int u = ((unsigned int)h) << 16;
  return __builtin_bit_cast(float, u);
}

#ifdef HAVE_ASYNC_LDS
typedef __attribute__((address_space(1))) v4i gv4i;   // global int4
typedef __attribute__((address_space(3))) v4i lv4i;   // LDS int4

__device__ __forceinline__ void async_cp16(const bf16_t* g, bf16_t* l) {
  __builtin_amdgcn_global_load_async_to_lds_b128(
      (gv4i*)(bf16_t*)g, (lv4i*)l, 0, 0);
}
__device__ __forceinline__ void wait_async0() {
#if __has_builtin(__builtin_amdgcn_s_wait_asynccnt)
  __builtin_amdgcn_s_wait_asynccnt(0);
#else
  asm volatile("s_wait_asynccnt 0x0" ::: "memory");
#endif
}
#endif

// ---- prep kernels ---------------------------------------------------------

// Wt[n][k] = bf16(W[k][n]); W is [K][N] row-major f32.
__global__ void k_transpose_bf(const float* __restrict__ W, bf16_t* __restrict__ Wt,
                               int K, int N) {
  long long idx = (long long)blockIdx.x * blockDim.x + threadIdx.x;
  if (idx >= (long long)K * N) return;
  int k = (int)(idx / N);
  int n = (int)(idx % N);
  Wt[(long long)n * K + k] = f2bf(W[idx]);
}

// W0xT[n][k] (k in [0,128)) = k<100 ? bf16(w0x[k][n]) : 0
__global__ void k_pad_w0x(const float* __restrict__ W, bf16_t* __restrict__ Wt) {
  int idx = blockIdx.x * blockDim.x + threadIdx.x;
  if (idx >= UNITS * EMBP) return;
  int n = idx / EMBP, k = idx % EMBP;
  Wt[idx] = (k < EMB) ? f2bf(W[(long long)k * UNITS + n]) : f2bf(0.0f);
}

// XE[t][b][k] = k<100 ? bf16(emb[tokens[b][t]][k]) : 0
__global__ void k_embed(const int* __restrict__ tokens, const float* __restrict__ emb,
                        bf16_t* __restrict__ XE) {
  long long idx = (long long)blockIdx.x * blockDim.x + threadIdx.x;
  if (idx >= (long long)SEQ * BATCH * EMBP) return;
  int k = (int)(idx % EMBP);
  int b = (int)((idx / EMBP) % BATCH);
  int t = (int)(idx / ((long long)EMBP * BATCH));
  bf16_t v = f2bf(0.0f);
  if (k < EMB) {
    int tok = tokens[b * SEQ + t];
    v = f2bf(emb[(long long)tok * EMB + k]);
  }
  XE[idx] = v;
}

__global__ void k_zero_bf16(bf16_t* __restrict__ p, long long n) {
  long long idx = (long long)blockIdx.x * blockDim.x + threadIdx.x;
  if (idx < n) p[idx] = f2bf(0.0f);
}

// ---- fused GEMM: out = act(X@Wx + H@Wh + bias) ----------------------------
// X: [512][ldx] bf16 (row-major), WxT: [2048][Kx] bf16 (transposed, contiguous K)
// H: [512][2048] bf16 or nullptr, WhT: [2048][Kh]
// out: [512][2048] bf16.  act: 0=tanh, 1=relu
__global__ __launch_bounds__(256) void k_gemm_rnn(
    const bf16_t* __restrict__ X, int ldx, int Kx,
    const bf16_t* __restrict__ WxT,
    const bf16_t* __restrict__ H, int Kh,
    const bf16_t* __restrict__ WhT,
    const float* __restrict__ bias,
    bf16_t* __restrict__ out, int act)
{
  __shared__ __align__(16) bf16_t sA[2][BM][LDS_K];   // 2 x 128 x (64 pad 72)
  __shared__ __align__(16) bf16_t sB[2][BN][LDS_K];   // 2 x 64  x (64 pad 72)

  const int tid   = threadIdx.x;
  const int lane  = tid & 31;
  const int wv    = tid >> 5;        // 0..7
  const int waveM = wv >> 1;         // 0..3 -> 32 rows each
  const int waveN = wv & 1;          // 0..1 -> 32 cols each
  const int hlf   = lane >> 4;       // 0/1
  const int l16   = lane & 15;

  const int m0 = blockIdx.y * BM;
  const int n0 = blockIdx.x * BN;

  v8f acc[2][2];
  #pragma unroll
  for (int i = 0; i < 2; ++i)
    #pragma unroll
    for (int j = 0; j < 2; ++j)
      acc[i][j] = v8f{0.f,0.f,0.f,0.f,0.f,0.f,0.f,0.f};

  // staging assignments: A 128 rows x 64 K (2 thr/row, 64B each = 4x b128)
  //                      B  64 rows x 64 K (4 thr/row, 32B each = 2x b128)
  const int arow  = tid >> 1;          // 0..127
  const int akoff = (tid & 1) * 32;    // 0 / 32
  const int brow  = tid >> 2;          // 0..63
  const int bkoff = (tid & 3) * 16;    // 0,16,32,48

  // issue stage loads for K range [kbase, kbase+64) into buffer `buf`
  auto issue = [&](const bf16_t* Am, int lda, const bf16_t* Bt, int ldb,
                   int kbase, int buf) {
    const bf16_t* asrc = Am + (long long)(m0 + arow) * lda + kbase + akoff;
    bf16_t*       adst = &sA[buf][arow][akoff];
    const bf16_t* bsrc = Bt + (long long)(n0 + brow) * ldb + kbase + bkoff;
    bf16_t*       bdst = &sB[buf][brow][bkoff];
#ifdef HAVE_ASYNC_LDS
    #pragma unroll
    for (int i = 0; i < 4; ++i) async_cp16(asrc + i * 8, adst + i * 8);
    #pragma unroll
    for (int i = 0; i < 2; ++i) async_cp16(bsrc + i * 8, bdst + i * 8);
#else
    uint4 a0 = ((const uint4*)asrc)[0];
    uint4 a1 = ((const uint4*)asrc)[1];
    uint4 a2 = ((const uint4*)asrc)[2];
    uint4 a3 = ((const uint4*)asrc)[3];
    uint4 b0 = ((const uint4*)bsrc)[0];
    uint4 b1 = ((const uint4*)bsrc)[1];
    ((uint4*)adst)[0] = a0;
    ((uint4*)adst)[1] = a1;
    ((uint4*)adst)[2] = a2;
    ((uint4*)adst)[3] = a3;
    ((uint4*)bdst)[0] = b0;
    ((uint4*)bdst)[1] = b1;
#endif
  };

  // consume one 64-K stage from buffer `buf`: 2 chunks x 4 WMMA
  // A frag (16-bit A 16x32): lanes 0-15 K={0..7,16..23}, lanes 16-31 K={8..15,24..31}
  // B frag:                  lanes 0-15 K={0..7,16..23}, lanes 16-31 K={8..15,24..31}
  auto compute = [&](int buf) {
    #pragma unroll
    for (int j = 0; j < 2; ++j) {
      const int kb = j * KC;
      v16bf afr[2], bfr[2];
      #pragma unroll
      for (int tm = 0; tm < 2; ++tm) {
        int r = waveM * 32 + tm * 16 + l16;
        v8bf lo = *(const v8bf*)&sA[buf][r][kb + hlf * 8];
        v8bf hi = *(const v8bf*)&sA[buf][r][kb + 16 + hlf * 8];
        afr[tm] = __builtin_shufflevector(lo, hi, 0,1,2,3,4,5,6,7,8,9,10,11,12,13,14,15);
      }
      #pragma unroll
      for (int tn = 0; tn < 2; ++tn) {
        int c = waveN * 32 + tn * 16 + l16;
        v8bf lo = *(const v8bf*)&sB[buf][c][kb + hlf * 8];
        v8bf hi = *(const v8bf*)&sB[buf][c][kb + 16 + hlf * 8];
        bfr[tn] = __builtin_shufflevector(lo, hi, 0,1,2,3,4,5,6,7,8,9,10,11,12,13,14,15);
      }
      #pragma unroll
      for (int tm = 0; tm < 2; ++tm)
        #pragma unroll
        for (int tn = 0; tn < 2; ++tn)
          acc[tm][tn] = __builtin_amdgcn_wmma_f32_16x16x32_bf16(
              false, afr[tm], false, bfr[tn], (short)0, acc[tm][tn], false, false);
    }
  };

  #pragma unroll
  for (int pass = 0; pass < 2; ++pass) {
    const bf16_t* Am = (pass == 0) ? X   : H;
    const bf16_t* Bt = (pass == 0) ? WxT : WhT;
    const int K      = (pass == 0) ? Kx  : Kh;
    const int lda    = (pass == 0) ? ldx : UNITS;
    if (Am == nullptr || K == 0) continue;
    const int nst = K / KSTEP;
    issue(Am, lda, Bt, K, 0, 0);
#ifdef HAVE_ASYNC_LDS
    wait_async0();
#endif
    __syncthreads();
    for (int s = 0; s < nst; ++s) {
      if (s + 1 < nst) issue(Am, lda, Bt, K, (s + 1) * KSTEP, (s + 1) & 1);
      compute(s & 1);
#ifdef HAVE_ASYNC_LDS
      wait_async0();     // next stage staged; overlapped with compute above
#endif
      __syncthreads();   // all waves done consuming buf (s&1) -> reusable
    }
  }

  // epilogue: bias + activation, store bf16
  // C/D layout: lane(0-15): N=l16, M=r; lane(16-31): N=l16, M=r+8
  #pragma unroll
  for (int tm = 0; tm < 2; ++tm)
    #pragma unroll
    for (int tn = 0; tn < 2; ++tn) {
      int nn = n0 + waveN * 32 + tn * 16 + l16;
      float bv = bias[nn];
      #pragma unroll
      for (int r = 0; r < 8; ++r) {
        int mm = m0 + waveM * 32 + tm * 16 + hlf * 8 + r;
        float v = acc[tm][tn][r] + bv;
        v = (act == 0) ? tanhf(v) : fmaxf(v, 0.0f);
        out[(long long)mm * UNITS + nn] = f2bf(v);
      }
    }
}

// ---- head: out[b] = sigmoid(hidden[b,:] . wd2 + bd2) ----------------------
__global__ __launch_bounds__(256) void k_head_dot(
    const bf16_t* __restrict__ hid, const float* __restrict__ wd2,
    const float* __restrict__ bd2, float* __restrict__ out)
{
  int wv = threadIdx.x >> 5, lane = threadIdx.x & 31;
  int row = blockIdx.x * 8 + wv;
  if (row >= BATCH) return;
  float s = 0.0f;
  for (int k = lane; k < UNITS; k += 32)
    s += bf2f(hid[(long long)row * UNITS + k]) * wd2[k];
  #pragma unroll
  for (int off = 16; off >= 1; off >>= 1)
    s += __shfl_xor(s, off, 32);
  if (lane == 0) out[row] = 1.0f / (1.0f + expf(-(s + bd2[0])));
}

// ---------------------------------------------------------------------------

extern "C" void kernel_launch(void* const* d_in, const int* in_sizes, int n_in,
                              void* d_out, int out_size, void* d_ws, size_t ws_size,
                              hipStream_t stream) {
  const int*   tokens = (const int*)  d_in[0];
  const float* emb    = (const float*)d_in[1];
  const float* w0x    = (const float*)d_in[2];
  const float* w0h    = (const float*)d_in[3];
  const float* b0     = (const float*)d_in[4];
  const float* w1x    = (const float*)d_in[5];
  const float* w1h    = (const float*)d_in[6];
  const float* b1     = (const float*)d_in[7];
  const float* wd1    = (const float*)d_in[8];
  const float* bd1    = (const float*)d_in[9];
  const float* wd2    = (const float*)d_in[10];
  const float* bd2    = (const float*)d_in[11];
  float* out = (float*)d_out;

  char* ws = (char*)d_ws;
  size_t off = 0;
  auto alloc = [&](size_t bytes) -> void* {
    void* p = ws + off;
    off = (off + bytes + 255) & ~(size_t)255;
    return p;
  };
  bf16_t* XE   = (bf16_t*)alloc((size_t)SEQ * BATCH * EMBP * 2);   // 10.0 MB
  bf16_t* W0XT = (bf16_t*)alloc((size_t)UNITS * EMBP * 2);
  bf16_t* W0HT = (bf16_t*)alloc((size_t)UNITS * UNITS * 2);
  bf16_t* W1XT = (bf16_t*)alloc((size_t)UNITS * UNITS * 2);
  bf16_t* W1HT = (bf16_t*)alloc((size_t)UNITS * UNITS * 2);
  bf16_t* WD1T = (bf16_t*)alloc((size_t)UNITS * UNITS * 2);
  bf16_t* H0A  = (bf16_t*)alloc((size_t)BATCH * UNITS * 2);        // ping/pong,
  bf16_t* H0B  = (bf16_t*)alloc((size_t)BATCH * UNITS * 2);        // contiguous
  bf16_t* H1A  = (bf16_t*)alloc((size_t)BATCH * UNITS * 2);
  bf16_t* H1B  = (bf16_t*)alloc((size_t)BATCH * UNITS * 2);
  bf16_t* HID  = (bf16_t*)alloc((size_t)BATCH * UNITS * 2);

  const int TPB = 256;
  long long nW = (long long)UNITS * UNITS;
  // weight prep (bf16, transposed to [N][K])
  k_transpose_bf<<<(int)((nW + TPB - 1) / TPB), TPB, 0, stream>>>(w0h, W0HT, UNITS, UNITS);
  k_transpose_bf<<<(int)((nW + TPB - 1) / TPB), TPB, 0, stream>>>(w1x, W1XT, UNITS, UNITS);
  k_transpose_bf<<<(int)((nW + TPB - 1) / TPB), TPB, 0, stream>>>(w1h, W1HT, UNITS, UNITS);
  k_transpose_bf<<<(int)((nW + TPB - 1) / TPB), TPB, 0, stream>>>(wd1, WD1T, UNITS, UNITS);
  k_pad_w0x<<<(UNITS * EMBP + TPB - 1) / TPB, TPB, 0, stream>>>(w0x, W0XT);
  long long nE = (long long)SEQ * BATCH * EMBP;
  k_embed<<<(int)((nE + TPB - 1) / TPB), TPB, 0, stream>>>(tokens, emb, XE);
  long long nH = 4LL * BATCH * UNITS;  // H0A..H1B contiguous
  k_zero_bf16<<<(int)((nH + TPB - 1) / TPB), TPB, 0, stream>>>(H0A, nH);

  dim3 grid(UNITS / BN, BATCH / BM);   // (32, 4)

  bf16_t *h0p = H0A, *h0n = H0B, *h1p = H1A, *h1n = H1B;
  for (int t = 0; t < SEQ; ++t) {
    // layer 0: h0' = tanh(x_t @ w0x + h0 @ w0h + b0)
    k_gemm_rnn<<<grid, TPB, 0, stream>>>(
        XE + (long long)t * BATCH * EMBP, EMBP, EMBP, W0XT,
        h0p, UNITS, W0HT, b0, h0n, /*tanh*/0);
    // layer 1: h1' = tanh(h0' @ w1x + h1 @ w1h + b1)
    k_gemm_rnn<<<grid, TPB, 0, stream>>>(
        h0n, UNITS, UNITS, W1XT,
        h1p, UNITS, W1HT, b1, h1n, /*tanh*/0);
    bf16_t* tmp;
    tmp = h0p; h0p = h0n; h0n = tmp;
    tmp = h1p; h1p = h1n; h1n = tmp;
  }
  // head dense + relu (no recurrent term)
  k_gemm_rnn<<<grid, TPB, 0, stream>>>(
      h1p, UNITS, UNITS, WD1T,
      nullptr, 0, W0HT, bd1, HID, /*relu*/1);
  // final dot + sigmoid
  k_head_dot<<<BATCH / 8, TPB, 0, stream>>>(HID, wd2, bd2, out);
}
